// Simple_GNN_27522150432987
// MI455X (gfx1250) — matrix operand order repaired and checked by reading, not compiled, not run on hardware
//
#include <hip/hip_runtime.h>

#define N_NODES 100000
#define N_EDGES 1600000
#define D_IN    128
#define D_HID   64

typedef __attribute__((ext_vector_type(16))) _Float16 v16h;
typedef __attribute__((ext_vector_type(8)))  float    v8f;

// ---------------- degree / normalization ----------------

__global__ void k_deg_init(float* __restrict__ deg) {
    int n = blockIdx.x * blockDim.x + threadIdx.x;
    if (n < N_NODES) deg[n] = 1.0f;   // self-loop contribution
}

__global__ void k_deg_count(const long long* __restrict__ ei, float* __restrict__ deg) {
    int e = blockIdx.x * blockDim.x + threadIdx.x;
    if (e < N_EDGES) atomicAdd(&deg[ei[N_EDGES + e]], 1.0f);
}

__global__ void k_deg_to_dis(float* __restrict__ deg) {
    int n = blockIdx.x * blockDim.x + threadIdx.x;
    if (n < N_NODES) {
        float d = deg[n];
        deg[n] = (d > 0.0f) ? rsqrtf(d) : 0.0f;   // in-place: deg -> deg^{-1/2}
    }
}

// ---------------- layer-1 GEMM: H = x @ W1 via v_wmma_f32_16x16x32_f16 ----------------
// One wave per 16-row M-tile; 4 N-tiles (D_HID=64) x 4 K-tiles (D_IN=128).

__global__ void k_gemm1_wmma(const float* __restrict__ x,
                             const float* __restrict__ W1,
                             float* __restrict__ H) {
    const int lane  = threadIdx.x & 31;
    const int wave  = threadIdx.x >> 5;
    const int mtile = blockIdx.x * (blockDim.x >> 5) + wave;
    const int m0    = mtile * 16;
    if (m0 >= N_NODES) return;          // wave-uniform: EXEC stays all-ones for WMMA

    const int lr = lane & 15;           // row within A-tile / col within B,C,D tile
    const int kh = lane >> 4;           // lane half selects K sub-range

    v8f cacc[4] = {};                   // 4 N-tile accumulators, 16x16 f32 each

    const float* xrow = x + (size_t)(m0 + lr) * D_IN;

#pragma unroll
    for (int kt = 0; kt < 4; ++kt) {
        // A fragment: 16x32 f16 (ISA 7.12.2 layout). Per lane: 8 VGPRs x 2 halves.
        v16h a;
#pragma unroll
        for (int v = 0; v < 8; ++v) {
            int kk = ((v < 4) ? 0 : 16) + kh * 8 + 2 * (v & 3);
            a[2 * v]     = (_Float16)xrow[kt * 32 + kk];
            a[2 * v + 1] = (_Float16)xrow[kt * 32 + kk + 1];
        }
#pragma unroll
        for (int nt = 0; nt < 4; ++nt) {
            // B fragment: 32x16 f16, column = lr, K = (e<8 ? 0 : 8) + kh*8 + e
            v16h b;
#pragma unroll
            for (int e = 0; e < 16; ++e) {
                int kk = ((e < 8) ? 0 : 8) + kh * 8 + e;
                b[e] = (_Float16)W1[(size_t)(kt * 32 + kk) * D_HID + nt * 16 + lr];
            }
            cacc[nt] = __builtin_amdgcn_wmma_f32_16x16x32_f16(
                /*neg_a=*/false, a, /*neg_b=*/false, b,
                /*c_mod=*/(short)0, cacc[nt], /*reuse_a=*/false, /*reuse_b=*/false);
        }
    }

    // D layout: VGPR r -> row (r + 8*kh), col lr
    float* hbase = H + (size_t)m0 * D_HID;
#pragma unroll
    for (int nt = 0; nt < 4; ++nt) {
#pragma unroll
        for (int r = 0; r < 8; ++r) {
            hbase[(size_t)(r + kh * 8) * D_HID + nt * 16 + lr] = cacc[nt][r];
        }
    }
}

// ---------------- layer-1 aggregation ----------------

// agg1[n,f] = H[n,f] * dis[n]^2   (self-loop term; also zero-initializes agg1)
__global__ void k_agg1_init(const float* __restrict__ H, const float* __restrict__ dis,
                            float* __restrict__ agg1) {
    int t = blockIdx.x * blockDim.x + threadIdx.x;
    if (t >= N_NODES * D_HID) return;
    float di = dis[t >> 6];
    agg1[t] = H[t] * di * di;
}

// per (edge, feature): agg1[dst,f] += H[src,f] * dis[src]*dis[dst]
__global__ void k_edge1(const long long* __restrict__ ei, const float* __restrict__ dis,
                        const float* __restrict__ H, float* __restrict__ agg1) {
    unsigned t = blockIdx.x * blockDim.x + threadIdx.x;
    if (t >= (unsigned)N_EDGES * D_HID) return;
    unsigned e = t >> 6;
    unsigned f = t & 63;
    long long s = ei[e];
    long long d = ei[N_EDGES + e];
    float w = dis[s] * dis[d];
    atomicAdd(&agg1[(size_t)d * D_HID + f], H[(size_t)s * D_HID + f] * w);
}

// h1 = relu(agg1 + b1), written back over H (x@W1 no longer needed)
__global__ void k_relu_bias(const float* __restrict__ agg1, const float* __restrict__ b1,
                            float* __restrict__ H) {
    int t = blockIdx.x * blockDim.x + threadIdx.x;
    if (t >= N_NODES * D_HID) return;
    float v = agg1[t] + b1[t & 63];
    H[t] = (v > 0.0f) ? v : 0.0f;
}

// ---------------- layer-2: z = h1 @ W2, out0 self-loop init ----------------

__global__ void k_layer2_mv(const float* __restrict__ H, const float* __restrict__ W2,
                            const float* __restrict__ dis,
                            float* __restrict__ z, float* __restrict__ out0) {
    int lane = threadIdx.x & 31;
    int node = (int)((blockIdx.x * blockDim.x + threadIdx.x) >> 5);
    if (node >= N_NODES) return;
    const float* hrow = H + (size_t)node * D_HID;
    float p = hrow[lane] * W2[lane] + hrow[lane + 32] * W2[lane + 32];
#pragma unroll
    for (int off = 16; off > 0; off >>= 1)
        p += __shfl_xor(p, off, 32);
    if (lane == 0) {
        z[node] = p;
        float di = dis[node];
        out0[node] = p * di * di;    // self-loop term of layer-2 aggregation
    }
}

__global__ void k_edge2(const long long* __restrict__ ei, const float* __restrict__ dis,
                        const float* __restrict__ z, float* __restrict__ out0) {
    int e = blockIdx.x * blockDim.x + threadIdx.x;
    if (e >= N_EDGES) return;
    long long s = ei[e];
    long long d = ei[N_EDGES + e];
    atomicAdd(&out0[d], z[s] * dis[s] * dis[d]);
}

// ---------------- final: bias + pad to [N, 128] ----------------

__global__ void k_final(const float* __restrict__ out0, const float* __restrict__ b2,
                        float* __restrict__ out) {
    int t = blockIdx.x * blockDim.x + threadIdx.x;
    if (t >= N_NODES * D_IN) return;
    int n = t >> 7;
    int j = t & 127;
    out[t] = (j == 0) ? (out0[n] + b2[0]) : 0.0f;
}

// ---------------- host launcher ----------------

extern "C" void kernel_launch(void* const* d_in, const int* in_sizes, int n_in,
                              void* d_out, int out_size, void* d_ws, size_t ws_size,
                              hipStream_t stream) {
    const float*     x  = (const float*)d_in[0];
    const long long* ei = (const long long*)d_in[1];   // int64 [2, E]: row0=src, row1=dst
    const float*     W1 = (const float*)d_in[2];
    const float*     b1 = (const float*)d_in[3];
    const float*     W2 = (const float*)d_in[4];
    const float*     b2 = (const float*)d_in[5];
    float*           out = (float*)d_out;

    // workspace carve-out (~52.5 MB total)
    float* H    = (float*)d_ws;                        // [N, 64]  x@W1, then h1
    float* agg1 = H + (size_t)N_NODES * D_HID;         // [N, 64]
    float* dis  = agg1 + (size_t)N_NODES * D_HID;      // [N]  deg -> deg^{-1/2}
    float* z    = dis + N_NODES;                       // [N]
    float* out0 = z + N_NODES;                         // [N]

    const int B = 256;

    // degree / normalization
    k_deg_init  <<<(N_NODES + B - 1) / B, B, 0, stream>>>(dis);
    k_deg_count <<<(N_EDGES + B - 1) / B, B, 0, stream>>>(ei, dis);
    k_deg_to_dis<<<(N_NODES + B - 1) / B, B, 0, stream>>>(dis);

    // layer-1 GEMM (WMMA): 6250 M-tiles, 8 waves per block
    const int mtiles = N_NODES / 16;                   // 100000 / 16 = 6250 exactly
    const int wavesPerBlock = B / 32;
    k_gemm1_wmma<<<(mtiles + wavesPerBlock - 1) / wavesPerBlock, B, 0, stream>>>(x, W1, H);

    // layer-1 aggregation
    k_agg1_init<<<(N_NODES * D_HID + B - 1) / B, B, 0, stream>>>(H, dis, agg1);
    {
        unsigned total = (unsigned)N_EDGES * D_HID;    // 102.4M threads
        k_edge1<<<(total + B - 1) / B, B, 0, stream>>>(ei, dis, H, agg1);
    }
    k_relu_bias<<<(N_NODES * D_HID + B - 1) / B, B, 0, stream>>>(agg1, b1, H);

    // layer-2
    k_layer2_mv<<<(N_NODES * 32 + B - 1) / B, B, 0, stream>>>(H, W2, dis, z, out0);
    k_edge2<<<(N_EDGES + B - 1) / B, B, 0, stream>>>(ei, dis, z, out0);

    // bias + pad
    k_final<<<(N_NODES * D_IN + B - 1) / B, B, 0, stream>>>(out0, b2, out);
}
